// micro_block_74629351735389
// MI455X (gfx1250) — matrix operand order repaired and checked by reading, not compiled, not run on hardware
//
#include <hip/hip_runtime.h>

// ---------------------------------------------------------------------------
// CDNA5 WMMA types (wave32, v_wmma_f32_16x16x32_bf16)
// ---------------------------------------------------------------------------
typedef __attribute__((ext_vector_type(16))) __bf16 v16bf;
typedef __attribute__((ext_vector_type(8)))  float  v8f;
typedef unsigned short bf16_t;

union Frag { unsigned u[8]; v16bf v; };

__device__ __forceinline__ bf16_t f2bf(float f) {
    unsigned u = __float_as_uint(f);
    u += 0x7fffu + ((u >> 16) & 1u);     // round-to-nearest-even
    return (bf16_t)(u >> 16);
}
__device__ __forceinline__ float bf2f(bf16_t h) {
    return __uint_as_float(((unsigned)h) << 16);
}

#if __has_builtin(__builtin_amdgcn_global_load_async_to_lds_b32)
#define HAVE_ASYNC_LDS 1
typedef __attribute__((address_space(1))) int as1_int;   // global src
typedef __attribute__((address_space(3))) int as3_int;   // LDS dst
#endif

#ifdef HAVE_ASYNC_LDS
#if __has_builtin(__builtin_amdgcn_s_wait_asynccnt)
#define WAIT_ASYNC() __builtin_amdgcn_s_wait_asynccnt(0)
#else
#define WAIT_ASYNC() asm volatile("s_wait_asynccnt 0" ::: "memory")
#endif
#else
#define WAIT_ASYNC()
#endif

// ---------------------------------------------------------------------------
// GEMM:  C[m,n] (+)= sum_k A[m,k] * B^T[n,k]      (bf16 io, f32 acc)
//   A: row-major bf16 [M x K], lda           (activations / graph matrix)
//   B: n-major bf16, Bel(k,n) = Bp[n*ldb+k]  (weights W[CO][CI] or x2t)
//   C: row-major bf16 [M x N]; optional fp32 bias[n], relu, beta accumulate
// 256 threads = 8 waves; tile 128x64, BK=32; M%128==0, N%64==0, K%32==0.
// Double-buffered LDS fed by ASYNCcnt-tracked global_load_async_to_lds_b32.
// ---------------------------------------------------------------------------
__global__ void __launch_bounds__(256)
gemm_wmma(const bf16_t* __restrict__ A, long lda, long strideAb,
          const bf16_t* __restrict__ Bp, long ldb, long strideBb,
          bf16_t* __restrict__ C, long ldc, long strideCb,
          const float* __restrict__ bias, int K, int relu, int beta)
{
    __shared__ bf16_t As[2][128 * 32];   // [m][k]
    __shared__ bf16_t Bs[2][64 * 32];    // [n][k]

    const int tid = threadIdx.x;
    const int m0  = blockIdx.x * 128;
    const int n0  = blockIdx.y * 64;
    const bf16_t* Ab = A  + (long)blockIdx.z * strideAb;
    const bf16_t* Bb = Bp + (long)blockIdx.z * strideBb;
    bf16_t*       Cb = C  + (long)blockIdx.z * strideCb;

    const int wave = tid >> 5, lane = tid & 31;
    const int wm = (wave & 3) * 32, wn = (wave >> 2) * 32;
    const int lm = lane & 15, lh = lane >> 4;

    // dword-granular tile loaders (K pairs contiguous in both operands)
    auto load_tiles = [&](int k0, int buf) {
        // A: 128x32 bf16 = 2048 dwords, 8 per thread
        for (int i = tid; i < 2048; i += 256) {
            const int r = i >> 4, kp = i & 15;
            const bf16_t* g = &Ab[(long)(m0 + r) * lda + (k0 + 2 * kp)];
#ifdef HAVE_ASYNC_LDS
            __builtin_amdgcn_global_load_async_to_lds_b32(
                (as1_int*)(unsigned long long)(size_t)g,
                (as3_int*)(unsigned)(size_t)&As[buf][i * 2], 0, 0);
#else
            *(unsigned*)&As[buf][i * 2] = *(const unsigned*)g;
#endif
        }
        // B: 64x32 bf16 = 1024 dwords, 4 per thread
        for (int i = tid; i < 1024; i += 256) {
            const int r = i >> 4, kp = i & 15;
            const bf16_t* g = &Bb[(long)(n0 + r) * ldb + (k0 + 2 * kp)];
#ifdef HAVE_ASYNC_LDS
            __builtin_amdgcn_global_load_async_to_lds_b32(
                (as1_int*)(unsigned long long)(size_t)g,
                (as3_int*)(unsigned)(size_t)&Bs[buf][i * 2], 0, 0);
#else
            *(unsigned*)&Bs[buf][i * 2] = *(const unsigned*)g;
#endif
        }
    };

    v8f acc[2][2] = {};
    const int nIter = K >> 5;

    load_tiles(0, 0);
    for (int it = 0; it < nIter; ++it) {
        const int cur = it & 1;
        WAIT_ASYNC();
        __syncthreads();
        if (it + 1 < nIter) load_tiles((it + 1) << 5, cur ^ 1);

        Frag fa[2], fb[2];
        #pragma unroll
        for (int t = 0; t < 2; ++t) {
            const int mrow = (wm + t * 16 + lm) * 32;
            const int nrow = (wn + t * 16 + lm) * 32;
            #pragma unroll
            for (int r = 0; r < 8; ++r) {
                // ISA 7.12.2: 16-bit A 16x32 (K jump at VGPR4, +8 per lane-half)
                const int kkA = 2 * r + ((r >= 4) ? 8 : 0) + 8 * lh;
                // 16-bit B 32x16 (same K order as A: jump at VGPR4, +8 per lane-half)
                const int kkB = 2 * r + ((r >= 4) ? 8 : 0) + 8 * lh;
                fa[t].u[r] = *(const unsigned*)&As[cur][mrow + kkA];
                fb[t].u[r] = *(const unsigned*)&Bs[cur][nrow + kkB];
            }
        }
        #pragma unroll
        for (int i = 0; i < 2; ++i)
            #pragma unroll
            for (int j = 0; j < 2; ++j)
                acc[i][j] = __builtin_amdgcn_wmma_f32_16x16x32_bf16(
                    false, fa[i].v, false, fb[j].v,
                    (short)0, acc[i][j], false, false);
        __syncthreads();
    }

    // Epilogue. D layout: lanes0-15 VGPRr->M=r, lanes16-31 -> M=r+8, N=lane%16
    #pragma unroll
    for (int i = 0; i < 2; ++i) {
        #pragma unroll
        for (int j = 0; j < 2; ++j) {
            const int nn = n0 + wn + j * 16 + lm;
            #pragma unroll
            for (int r = 0; r < 8; ++r) {
                const long mm = m0 + wm + i * 16 + r + 8 * lh;
                const long ci = mm * ldc + nn;
                float vv = acc[i][j][r];
                if (beta) vv += bf2f(Cb[ci]);
                if (bias) vv += bias[nn];
                if (relu) vv = fmaxf(vv, 0.f);
                Cb[ci] = f2bf(vv);
            }
        }
    }
}

// ---------------------------------------------------------------------------
// Per-batch 1024x1280 -> 1280x1024 bf16 transpose (tiled through LDS)
// ---------------------------------------------------------------------------
__global__ void __launch_bounds__(256)
transpose_kernel(const bf16_t* __restrict__ src, bf16_t* __restrict__ dst)
{
    __shared__ bf16_t tile[32][33];
    const long base = (long)blockIdx.z * 1024 * 1280;
    const int v0 = blockIdx.x * 32;          // 1024 / 32
    const int j0 = blockIdx.y * 32;          // 1280 / 32
    const int tx = threadIdx.x & 31, ty = threadIdx.x >> 5;
    for (int rr = ty; rr < 32; rr += 8)
        tile[rr][tx] = src[base + (long)(v0 + rr) * 1280 + j0 + tx];
    __syncthreads();
    for (int rr = ty; rr < 32; rr += 8)
        dst[base + (long)(j0 + rr) * 1024 + v0 + tx] = tile[tx][rr];
}

// ---------------------------------------------------------------------------
// fp32 -> bf16 bulk conversion (graph matrices, weights)
// ---------------------------------------------------------------------------
__global__ void __launch_bounds__(256)
cvt_bf16_kernel(const float* __restrict__ src, bf16_t* __restrict__ dst, int n)
{
    int i = blockIdx.x * 256 + threadIdx.x;
    if (i < n) dst[i] = f2bf(src[i]);
}

// ---------------------------------------------------------------------------
// TC: dilated (1x3, dil=2) valid conv + bias + relu, plus pack tem_embedding.
// x[B,32,1024,24](f32) -> xe[b,n,t,0:64]=relu(conv), xe[...,64:96]=tem[...,t+4]
// ---------------------------------------------------------------------------
__global__ void __launch_bounds__(64)
tc_embed_kernel(const float* __restrict__ x, const float* __restrict__ tem,
                const float* __restrict__ w, const float* __restrict__ b,
                bf16_t* __restrict__ xe)
{
    __shared__ float xs[32 * 24];
    __shared__ float wsm[64 * 96];
    const int bn = blockIdx.x;
    const int n  = bn & 1023;
    const int bb = bn >> 10;
    const int tid = threadIdx.x;

    for (int i = tid; i < 32 * 24; i += 64) {
        int ci = i / 24, t = i % 24;
        xs[i] = x[(((long)bb * 32 + ci) * 1024 + n) * 24 + t];
    }
    for (int i = tid; i < 64 * 96; i += 64) wsm[i] = w[i];
    __syncthreads();

    const int co = tid;
    const float bc = b[co];
    const long base = (long)bn * 20 * 96;
    for (int t = 0; t < 20; ++t) {
        float acc = bc;
        #pragma unroll
        for (int ci = 0; ci < 32; ++ci) {
            const float* wr = &wsm[co * 96 + ci * 3];
            const float* xr = &xs[ci * 24 + t];
            acc += wr[0] * xr[0] + wr[1] * xr[2] + wr[2] * xr[4];
        }
        xe[base + t * 96 + co] = f2bf(fmaxf(acc, 0.f));
    }
    for (int i = tid; i < 32 * 20; i += 64) {
        int ce = i / 20, t = i % 20;
        xe[base + t * 96 + 64 + ce] =
            f2bf(tem[(((long)bb * 32 + ce) * 1024 + n) * 24 + (t + 4)]);
    }
}

// ---------------------------------------------------------------------------
// Masked multi-head temporal attention. q,k,v,o: [B,N,20,64] bf16 (8 heads x 8)
// ---------------------------------------------------------------------------
__global__ void __launch_bounds__(64)
tatt_kernel(const bf16_t* __restrict__ q, const bf16_t* __restrict__ k,
            const bf16_t* __restrict__ v, bf16_t* __restrict__ o)
{
    __shared__ float qs[20 * 64], ks[20 * 64], vs[20 * 64];
    const long base = (long)blockIdx.x * 20 * 64;
    const int tid = threadIdx.x;
    for (int i = tid; i < 1280; i += 64) {
        qs[i] = bf2f(q[base + i]); ks[i] = bf2f(k[base + i]); vs[i] = bf2f(v[base + i]);
    }
    __syncthreads();

    const float scale = 0.35355339059327373f;  // 1/sqrt(8)
    for (int r = tid; r < 160; r += 64) {
        const int h = r / 20, t = r % 20, d0 = h * 8;
        float p[20];
        float mx = -3.4e38f;
        for (int s = 0; s < 20; ++s) {
            float sc;
            if (s <= t) {
                sc = 0.f;
                #pragma unroll
                for (int d = 0; d < 8; ++d) sc += qs[t * 64 + d0 + d] * ks[s * 64 + d0 + d];
                sc *= scale;
            } else {
                sc = -32767.0f;
            }
            p[s] = sc; mx = fmaxf(mx, sc);
        }
        float sum = 0.f;
        for (int s = 0; s < 20; ++s) { p[s] = __expf(p[s] - mx); sum += p[s]; }
        const float inv = 1.f / sum;
        #pragma unroll
        for (int d = 0; d < 8; ++d) {
            float acc = 0.f;
            for (int s = 0; s < 20; ++s) acc += p[s] * vs[s * 64 + d0 + d];
            o[base + t * 64 + d0 + d] = f2bf(acc * inv);
        }
    }
}

// ---------------------------------------------------------------------------
// micro_fusion: a = softmax_g(<hh_g,u>); out[b,c,n,t] = sum_g a_g*hh_g  (NCHW f32)
// ---------------------------------------------------------------------------
__global__ void __launch_bounds__(256)
fuse_kernel(const bf16_t* __restrict__ hh0, const bf16_t* __restrict__ hh1,
            const bf16_t* __restrict__ u, float* __restrict__ out, int S)
{
    const int s = blockIdx.x * 256 + threadIdx.x;
    if (s >= S) return;
    const long b64 = (long)s * 64;
    float g0[64], g1[64];
    float d0 = 0.f, d1 = 0.f;
    for (int c = 0; c < 64; ++c) {
        g0[c] = bf2f(hh0[b64 + c]);
        g1[c] = bf2f(hh1[b64 + c]);
        const float uc = bf2f(u[b64 + c]);
        d0 += g0[c] * uc;
        d1 += g1[c] * uc;
    }
    const float m  = fmaxf(d0, d1);
    const float e0 = __expf(d0 - m), e1 = __expf(d1 - m);
    const float inv = 1.f / (e0 + e1);
    const float a0 = e0 * inv, a1 = e1 * inv;
    const int t = s % 20, n = (s / 20) & 1023, bb = s / (20 * 1024);
    for (int c = 0; c < 64; ++c) {
        out[(((long)bb * 64 + c) * 1024 + n) * 20 + t] = a0 * g0[c] + a1 * g1[c];
    }
}

// ---------------------------------------------------------------------------
// Orchestration
// ---------------------------------------------------------------------------
extern "C" void kernel_launch(void* const* d_in, const int* in_sizes, int n_in,
                              void* d_out, int out_size, void* d_ws, size_t ws_size,
                              hipStream_t stream)
{
    (void)in_sizes; (void)n_in; (void)out_size; (void)ws_size;
    const float* x    = (const float*)d_in[0];
    const float* tem  = (const float*)d_in[1];
    const float* A0   = (const float*)d_in[2];
    const float* A1   = (const float*)d_in[3];
    const float* A2   = (const float*)d_in[4];
    const float* A3   = (const float*)d_in[5];
    const float* w_tc = (const float*)d_in[6];
    const float* b_tc = (const float*)d_in[7];
    const float* wq   = (const float*)d_in[8];
    const float* bq   = (const float*)d_in[9];
    const float* wk   = (const float*)d_in[10];
    const float* bk   = (const float*)d_in[11];
    const float* wv   = (const float*)d_in[12];
    const float* bv   = (const float*)d_in[13];
    const float* wo   = (const float*)d_in[14];
    const float* bo   = (const float*)d_in[15];
    const float* wg0  = (const float*)d_in[16];
    const float* bg0  = (const float*)d_in[17];
    const float* wg1  = (const float*)d_in[18];
    const float* bg1  = (const float*)d_in[19];
    const float* wadp = (const float*)d_in[20];
    const float* badp = (const float*)d_in[21];
    const float* w0f  = (const float*)d_in[22];
    float* out = (float*)d_out;

    const long S  = 16L * 1024 * 20;    // 327680 spatial sites
    const long Uh = S * 64;             // bf16 elements per [S,64] buffer

    // Workspace carving (bf16) with reuse. Peak ~ 6.5*Uh + graphs ~ 290 MB.
    bf16_t* wsb  = (bf16_t*)d_ws;
    bf16_t* xe   = wsb;                  // [S,96]  : [0, 1.5Uh)
    bf16_t* qb   = wsb + (Uh * 3) / 2;   // [1.5, 2.5)
    bf16_t* kbuf = qb + Uh;              // [2.5, 3.5)
    bf16_t* vbuf = kbuf + Uh;            // [3.5, 4.5)
    bf16_t* h1   = vbuf + Uh;            // [4.5, 5.5)
    bf16_t* gA   = h1 + Uh;              // 4 x 1024*1024 bf16 graph matrices
    bf16_t* attn = wsb;                  // reuse xe region (dead after qkv)
    bf16_t* x2   = qb;                   // reuse q (dead after attention)
    bf16_t* nc0  = kbuf;                 // reuse k
    bf16_t* nc1  = vbuf;                 // reuse v
    bf16_t* h0   = wsb;                  // reuse attn (dead after wo GEMM)
    bf16_t* ub   = qb;                   // reuse x2 (dead after h1 slice-0 GEMM)
    bf16_t* hh0  = kbuf;                 // reuse nc2 slot (dead after h1)
    bf16_t* hh1  = vbuf;                 // reuse nc3 slot

    // bf16 weight copies + transposed activations
    bf16_t* wqh   = gA + 4L * 1024 * 1024;
    bf16_t* wkh   = wqh + 64 * 96;
    bf16_t* wvh   = wkh + 64 * 96;
    bf16_t* woh   = wvh + 64 * 96;
    bf16_t* wg0h  = woh + 64 * 64;
    bf16_t* wg1h  = wg0h + 64 * 192;
    bf16_t* wadph = wg1h + 64 * 192;
    bf16_t* w0fh  = wadph + 64 * 128;
    bf16_t* x2t   = w0fh + 64 * 64;      // [B][1280][1024] (Uh elements)

    const dim3 blk(256);
    const dim3 gS((unsigned)(S / 128), 1, 1);     // conv1x1 grids (M=S)
    const dim3 gNC(1024 / 128, 1280 / 64, 16);    // nconv grids (per batch)
    const long XB = 1024L * 1280;                 // per-batch stride

    // 0) convert graphs + weights to bf16
    const int NN = 1024 * 1024;
    cvt_bf16_kernel<<<(NN + 255) / 256, 256, 0, stream>>>(A0, gA,           NN);
    cvt_bf16_kernel<<<(NN + 255) / 256, 256, 0, stream>>>(A1, gA + 1L * NN, NN);
    cvt_bf16_kernel<<<(NN + 255) / 256, 256, 0, stream>>>(A2, gA + 2L * NN, NN);
    cvt_bf16_kernel<<<(NN + 255) / 256, 256, 0, stream>>>(A3, gA + 3L * NN, NN);
    cvt_bf16_kernel<<<(64 * 96 + 255) / 256, 256, 0, stream>>>(wq, wqh, 64 * 96);
    cvt_bf16_kernel<<<(64 * 96 + 255) / 256, 256, 0, stream>>>(wk, wkh, 64 * 96);
    cvt_bf16_kernel<<<(64 * 96 + 255) / 256, 256, 0, stream>>>(wv, wvh, 64 * 96);
    cvt_bf16_kernel<<<(64 * 64 + 255) / 256, 256, 0, stream>>>(wo, woh, 64 * 64);
    cvt_bf16_kernel<<<(64 * 192 + 255) / 256, 256, 0, stream>>>(wg0, wg0h, 64 * 192);
    cvt_bf16_kernel<<<(64 * 192 + 255) / 256, 256, 0, stream>>>(wg1, wg1h, 64 * 192);
    cvt_bf16_kernel<<<(64 * 128 + 255) / 256, 256, 0, stream>>>(wadp, wadph, 64 * 128);
    cvt_bf16_kernel<<<(64 * 64 + 255) / 256, 256, 0, stream>>>(w0f, w0fh, 64 * 64);

    // 1) TC conv + tem pack -> xe [S,96]
    tc_embed_kernel<<<16 * 1024, 64, 0, stream>>>(x, tem, w_tc, b_tc, xe);

    // 2) q/k/v = xe @ W^T + b    (K=96)
    gemm_wmma<<<gS, blk, 0, stream>>>(xe, 96, 0, wqh, 96, 0, qb,   64, 0, bq, 96, 0, 0);
    gemm_wmma<<<gS, blk, 0, stream>>>(xe, 96, 0, wkh, 96, 0, kbuf, 64, 0, bk, 96, 0, 0);
    gemm_wmma<<<gS, blk, 0, stream>>>(xe, 96, 0, wvh, 96, 0, vbuf, 64, 0, bv, 96, 0, 0);

    // 3) masked temporal attention -> attn [S,64]
    tatt_kernel<<<16 * 1024, 64, 0, stream>>>(qb, kbuf, vbuf, attn);

    // 4) x2 = relu(attn @ wo^T + bo), then x2t = per-batch transpose
    gemm_wmma<<<gS, blk, 0, stream>>>(attn, 64, 0, woh, 64, 0, x2, 64, 0, bo, 64, 1, 0);
    transpose_kernel<<<dim3(32, 40, 16), 256, 0, stream>>>(x2, x2t);

    // 5) graph diffusion nc_g[b] = A_g @ x2[b]   (M=K=1024, N=1280)
    gemm_wmma<<<gNC, blk, 0, stream>>>(gA,           1024, 0, x2t, 1024, XB, nc0, 1280, XB, nullptr, 1024, 0, 0);
    gemm_wmma<<<gNC, blk, 0, stream>>>(gA + 1L * NN, 1024, 0, x2t, 1024, XB, nc1, 1280, XB, nullptr, 1024, 0, 0);

    // 6) h0 = relu([x2|nc0|nc1] @ wg0^T + bg0) via accumulating GEMMs
    gemm_wmma<<<gS, blk, 0, stream>>>(x2,  64, 0, wg0h,       192, 0, h0, 64, 0, nullptr, 64, 0, 0);
    gemm_wmma<<<gS, blk, 0, stream>>>(nc0, 64, 0, wg0h + 64,  192, 0, h0, 64, 0, nullptr, 64, 0, 1);
    gemm_wmma<<<gS, blk, 0, stream>>>(nc1, 64, 0, wg0h + 128, 192, 0, h0, 64, 0, bg0,     64, 1, 1);

    // 7) nc2/nc3 into the nc0/nc1 slots, then h1
    gemm_wmma<<<gNC, blk, 0, stream>>>(gA + 2L * NN, 1024, 0, x2t, 1024, XB, nc0, 1280, XB, nullptr, 1024, 0, 0);
    gemm_wmma<<<gNC, blk, 0, stream>>>(gA + 3L * NN, 1024, 0, x2t, 1024, XB, nc1, 1280, XB, nullptr, 1024, 0, 0);
    gemm_wmma<<<gS, blk, 0, stream>>>(x2,  64, 0, wg1h,       192, 0, h1, 64, 0, nullptr, 64, 0, 0);
    gemm_wmma<<<gS, blk, 0, stream>>>(nc0, 64, 0, wg1h + 64,  192, 0, h1, 64, 0, nullptr, 64, 0, 1);
    gemm_wmma<<<gS, blk, 0, stream>>>(nc1, 64, 0, wg1h + 128, 192, 0, h1, 64, 0, bg1,     64, 1, 1);

    // 8) u = [h0|h1] @ w_adp^T + b_adp
    gemm_wmma<<<gS, blk, 0, stream>>>(h0, 64, 0, wadph,      128, 0, ub, 64, 0, nullptr, 64, 0, 0);
    gemm_wmma<<<gS, blk, 0, stream>>>(h1, 64, 0, wadph + 64, 128, 0, ub, 64, 0, badp,    64, 0, 1);

    // 9) hh_g = h_g @ w0f^T
    gemm_wmma<<<gS, blk, 0, stream>>>(h0, 64, 0, w0fh, 64, 0, hh0, 64, 0, nullptr, 64, 0, 0);
    gemm_wmma<<<gS, blk, 0, stream>>>(h1, 64, 0, w0fh, 64, 0, hh1, 64, 0, nullptr, 64, 0, 0);

    // 10) graph-softmax fusion -> NCHW fp32 output
    fuse_kernel<<<(unsigned)((S + 255) / 256), 256, 0, stream>>>(hh0, hh1, ub, out, (int)S);
}
